// NPI_80058190397602
// MI455X (gfx1250) — compile-verified, matrix operand order checked by reading, not run on hardware
//
#include <hip/hip_runtime.h>
#include <hip/hip_bf16.h>
#include <math.h>

// ---------------------------------------------------------------------------
// CDNA5 (gfx1250) NPI forward. wave32; all matmul-shaped work goes through
// v_wmma_f32_16x16x32_f16 (batch==16 == WMMA M). GEMM operands are staged
// into LDS with GLOBAL_LOAD_ASYNC_TO_LDS (ASYNCcnt) when available; f32->f16
// conversion happens in the LDS->VGPR fragment gather. Accumulation is f32.
// ---------------------------------------------------------------------------

typedef __attribute__((ext_vector_type(16))) _Float16 v16h;
typedef __attribute__((ext_vector_type(8)))  float    v8f;
typedef __attribute__((ext_vector_type(4)))  int      v4i;

#define PADC 3
#define NUC  30
#define NBC  10
#define PLENC 6
#define BATCH 16

#if __has_builtin(__builtin_amdgcn_global_load_async_to_lds_b128)
#define HAS_ASYNC_LDS 1
#else
#define HAS_ASYNC_LDS 0
#endif

#if HAS_ASYNC_LDS
// builtin signature (pinned by probe diagnostics): arg0 = v4i* in AS(1),
// arg1 = v4i* in AS(3), then imm offset, imm cpol.
#define ASYNC_B128(gptr, lptr)                                              \
  __builtin_amdgcn_global_load_async_to_lds_b128(                           \
      (__attribute__((address_space(1))) v4i*)(void*)(gptr),                \
      (__attribute__((address_space(3))) v4i*)(lptr), 0, 0)

__device__ __forceinline__ void wait_asynccnt0() {
#if __has_builtin(__builtin_amdgcn_s_wait_asynccnt)
  __builtin_amdgcn_s_wait_asynccnt(0);
#else
  asm volatile("s_wait_asynccnt 0x0" ::: "memory");
#endif
}
#endif

__device__ __forceinline__ float sigmf(float x) { return 1.0f / (1.0f + expf(-x)); }

// Gather a 16x16x32 WMMA operand fragment from a 16-row f16 LDS tile
// (row stride 32).  Per ISA 7.12.2 (16-bit A 16x32): lanes 0-15 hold rows
// 0..15 with K = {0..7,16..23}; lanes 16-31 the same rows, K = {8..15,24..31}.
__device__ __forceinline__ v16h frag16(const _Float16* t) {
  const int lane = threadIdx.x & 31;
  const int r    = lane & 15;
  const int hs   = (lane >> 4) & 1;
  v16h f;
#pragma unroll
  for (int g = 0; g < 8; ++g) {
    const int k0 = ((g < 4) ? (g << 1) : (16 + ((g - 4) << 1))) + (hs << 3);
    f[2 * g]     = t[r * 32 + k0];
    f[2 * g + 1] = t[r * 32 + k0 + 1];
  }
  return f;
}

// Same gather from a 16x64 f32 LDS tile (async-staged raw f32), converting to
// f16 on the fly; col0 selects the 32-wide K chunk.
__device__ __forceinline__ v16h fragf(const float* t, int col0) {
  const int lane = threadIdx.x & 31;
  const int r    = lane & 15;
  const int hs   = (lane >> 4) & 1;
  v16h f;
#pragma unroll
  for (int g = 0; g < 8; ++g) {
    const int k0 = ((g < 4) ? (g << 1) : (16 + ((g - 4) << 1))) + (hs << 3);
    f[2 * g]     = (_Float16)t[r * 64 + col0 + k0];
    f[2 * g + 1] = (_Float16)t[r * 64 + col0 + k0 + 1];
  }
  return f;
}

// ---------------------------------------------------------------------------
// Generic M=16 GEMM:  C[16,N] (+)= act( A[16,K] @ W[N,K]^T + bias )
// 4 waves per block share the A tile; each wave owns a 16-wide N slice.
// K-unrolled by 64: two WMMAs per barrier interval. Full interior tiles are
// staged with async global->LDS b128 copies (raw f32), ragged tiles fall back
// to scalar staging with zero fill.
// ---------------------------------------------------------------------------
__global__ void __launch_bounds__(128)
gemm16_wmma(const float* __restrict__ A, int lda,
            const float* __restrict__ W,
            const float* __restrict__ bias,
            float* __restrict__ C, int ldc,
            int N, int K, int accum, int relu) {
  __shared__ float Af[16 * 64];
  __shared__ float Bf[4][16 * 64];
  const int wave = threadIdx.x >> 5;
  const int lane = threadIdx.x & 31;
  const int n0   = blockIdx.x * 64 + wave * 16;
  v8f acc = {};
  for (int kt = 0; kt < K; kt += 64) {
#if HAS_ASYNC_LDS
    const bool fullA = (kt + 64 <= K);
    const bool fullB = fullA && (n0 + 16 <= N);
    if (fullA) {
      if (wave == 0) {  // 4KB A tile: 8 x (32 lanes x b128)
#pragma unroll
        for (int j = 0; j < 8; ++j) {
          const int c = j * 32 + lane, r = c >> 4, seg = c & 15;
          const float* g = A + (size_t)r * lda + kt + seg * 4;
          ASYNC_B128(g, &Af[r * 64 + seg * 4]);
        }
      }
    } else {
      for (int e = (int)threadIdx.x; e < 1024; e += 128) {
        const int r = e >> 6, kk = e & 63, k = kt + kk;
        Af[e] = (k < K) ? A[(size_t)r * lda + k] : 0.0f;
      }
    }
    if (fullB) {
#pragma unroll
      for (int j = 0; j < 8; ++j) {
        const int c = j * 32 + lane, r = c >> 4, seg = c & 15;
        const float* g = W + (size_t)(n0 + r) * K + kt + seg * 4;
        ASYNC_B128(g, &Bf[wave][r * 64 + seg * 4]);
      }
    } else {
      for (int e = lane; e < 1024; e += 32) {
        const int r = e >> 6, kk = e & 63, k = kt + kk, n = n0 + r;
        Bf[wave][e] = (n < N && k < K) ? W[(size_t)n * K + k] : 0.0f;
      }
    }
    wait_asynccnt0();  // ASYNCcnt is per-wave; barrier below publishes to WG
#else
    for (int e = (int)threadIdx.x; e < 1024; e += 128) {
      const int r = e >> 6, kk = e & 63, k = kt + kk;
      Af[e] = (k < K) ? A[(size_t)r * lda + k] : 0.0f;
    }
    for (int e = lane; e < 1024; e += 32) {
      const int r = e >> 6, kk = e & 63, k = kt + kk, n = n0 + r;
      Bf[wave][e] = (n < N && k < K) ? W[(size_t)n * K + k] : 0.0f;
    }
#endif
    __syncthreads();
    acc = __builtin_amdgcn_wmma_f32_16x16x32_f16(false, fragf(Af, 0), false,
                                                 fragf(Bf[wave], 0), (short)0,
                                                 acc, false, false);
    acc = __builtin_amdgcn_wmma_f32_16x16x32_f16(false, fragf(Af, 32), false,
                                                 fragf(Bf[wave], 32), (short)0,
                                                 acc, false, false);
    __syncthreads();
    if (kt + 64 < K)  // global_prefetch_b8
      __builtin_prefetch(&A[(size_t)(lane & 15) * lda + kt + 64], 0, 0);
  }
  const int n = n0 + (lane & 15);
  if (n < N) {
    const int mb = (lane >> 4) << 3;  // C layout: VGPR j -> M = mb + j, lane -> N
    const float bv = bias ? bias[n] : 0.0f;
#pragma unroll
    for (int j = 0; j < 8; ++j) {
      const int m = mb + j;
      float v = acc[j] + bv;
      if (accum) v += C[(size_t)m * ldc + n];
      if (relu)  v = fmaxf(v, 0.0f);
      C[(size_t)m * ldc + n] = v;
    }
  }
}

// ---------------------------------------------------------------------------
// Implicit-GEMM NCHW conv.  16 output pixels x 64 output channels per block.
// Optional 2-input channel concat (in0: Cin0 channels, in1: Cin-Cin0), optional
// per-sample weight bank (vmapped params) picked from prog[b], fused
// bias/residual/relu epilogue.
// ---------------------------------------------------------------------------
__global__ void __launch_bounds__(128)
conv_wmma(const float* __restrict__ in0, const float* __restrict__ in1,
          int Cin0, int Cin,
          const float* __restrict__ Wt, const float* __restrict__ Bs,
          const int* __restrict__ prog, int step, int idx_off, int idx_max,
          const float* __restrict__ res, float* __restrict__ out,
          int H, int Wd, int OH, int OW, int Cout,
          int stride, int pad, int ksz, int relu) {
  __shared__ _Float16 Ah[16 * 32];
  __shared__ _Float16 Bh[4][16 * 32];
  const int wave = threadIdx.x >> 5;
  const int lane = threadIdx.x & 31;
  const int b    = blockIdx.z;
  const int OHW  = OH * OW;
  const int pix0 = blockIdx.x * 16;
  const int co0  = blockIdx.y * 64 + wave * 16;
  const int kpix = ksz * ksz;
  const int K    = Cin * kpix;

  const float* wbase = Wt;
  const float* bbase = Bs;
  if (prog) {  // per-sample weight selection (unary/binary banks)
    int p = prog[b * PLENC + step] - idx_off;
    p = p < 0 ? 0 : (p > idx_max ? idx_max : p);
    wbase += (size_t)p * Cout * K;
    bbase += (size_t)p * Cout;
  }
  const float* i0 = in0 + (size_t)b * Cin0 * H * Wd;
  const float* i1 = in1 ? in1 + (size_t)b * (Cin - Cin0) * H * Wd : nullptr;

  v8f acc = {};
  for (int kt = 0; kt < K; kt += 32) {
    for (int e = (int)threadIdx.x; e < 512; e += 128) {
      int r = e >> 5, kk = e & 31, k = kt + kk;
      float v = 0.0f;
      const int pix = pix0 + r;
      if (k < K && pix < OHW) {
        const int ci = k / kpix;
        const int rr = k - ci * kpix;
        const int kh = rr / ksz, kw = rr - kh * ksz;
        const int oh = pix / OW, ow = pix - oh * OW;
        const int ih = oh * stride - pad + kh;
        const int iw = ow * stride - pad + kw;
        if (ih >= 0 && ih < H && iw >= 0 && iw < Wd) {
          const float* src = (ci < Cin0) ? (i0 + (size_t)ci * H * Wd)
                                         : (i1 + (size_t)(ci - Cin0) * H * Wd);
          v = src[ih * Wd + iw];
        }
      }
      Ah[e] = (_Float16)v;
    }
    for (int e = lane; e < 512; e += 32) {
      int r = e >> 5, kk = e & 31, k = kt + kk, co = co0 + r;
      Bh[wave][e] = (co < Cout && k < K) ? (_Float16)wbase[(size_t)co * K + k]
                                         : (_Float16)0.0f;
    }
    __syncthreads();
    v16h a  = frag16(Ah);
    v16h bb = frag16(Bh[wave]);
    acc = __builtin_amdgcn_wmma_f32_16x16x32_f16(false, a, false, bb, (short)0,
                                                 acc, false, false);
    __syncthreads();
  }
  const int co = co0 + (lane & 15);
  if (co < Cout) {
    float* ob        = out + (size_t)b * Cout * OHW + (size_t)co * OHW;
    const float* rb  = res ? res + (size_t)b * Cout * OHW + (size_t)co * OHW : nullptr;
    const float bv   = bbase[co];
    const int mb     = (lane >> 4) << 3;
#pragma unroll
    for (int j = 0; j < 8; ++j) {
      const int pix = pix0 + mb + j;
      if (pix < OHW) {
        float v = acc[j] + bv;
        if (rb)   v += rb[pix];
        if (relu) v = fmaxf(v, 0.0f);
        ob[pix] = v;
      }
    }
  }
}

// ------------------------------ pointwise ----------------------------------

__global__ void zero_k(float* p, int n) {
  int i = blockIdx.x * blockDim.x + threadIdx.x;
  if (i < n) p[i] = 0.0f;
}
__global__ void copy_k(const float* a, float* b, int n) {
  int i = blockIdx.x * blockDim.x + threadIdx.x;
  if (i < n) b[i] = a[i];
}
__global__ void embed_k(const int* __restrict__ q, const float* __restrict__ tab,
                        float* __restrict__ out) {
  int i = blockIdx.x * blockDim.x + threadIdx.x;
  if (i >= BATCH * 20 * 128) return;
  const int k = i & 127;
  const int t = (i >> 7) % 20;
  const int b = i / (20 * 128);
  out[i] = tab[(size_t)q[b * 20 + t] * 128 + k];
}
__global__ void lstm_gates_k(const float* __restrict__ z, float* __restrict__ h,
                             float* __restrict__ c, float* __restrict__ ys) {
  int i = blockIdx.x * blockDim.x + threadIdx.x;
  if (i >= BATCH * 256) return;
  const int b = i >> 8, j = i & 255;
  const float* zb = z + (size_t)b * 1024;
  const float ig = sigmf(zb[j]);
  const float fg = sigmf(zb[256 + j]);
  const float gg = tanhf(zb[512 + j]);
  const float og = sigmf(zb[768 + j]);
  const float cc = fg * c[i] + ig * gg;
  const float hh = og * tanhf(cc);
  c[i] = cc;
  h[i] = hh;
  if (ys) ys[i] = hh;
}
__global__ void rhn_pw_k(const float* __restrict__ ph, const float* __restrict__ pt,
                         float* __restrict__ s) {
  int i = blockIdx.x * blockDim.x + threadIdx.x;
  if (i >= BATCH * 256) return;
  const float hh = tanhf(ph[i]);
  const float tt = sigmf(pt[i]);
  s[i] = hh * tt + s[i] * (1.0f - tt);
}
__global__ void maxpool2_k(const float* __restrict__ x, float* __restrict__ y,
                           int H, int W, int total) {
  int i = blockIdx.x * blockDim.x + threadIdx.x;
  if (i >= total) return;
  const int ow2 = W >> 1, oh2 = H >> 1;
  const int ow = i % ow2;
  int t = i / ow2;
  const int oh = t % oh2;
  t /= oh2;  // t = b*C + c
  const float* p = x + (size_t)t * H * W;
  const int ih = oh * 2, iw = ow * 2;
  y[i] = fmaxf(fmaxf(p[ih * W + iw], p[ih * W + iw + 1]),
               fmaxf(p[(ih + 1) * W + iw], p[(ih + 1) * W + iw + 1]));
}
__global__ void instnorm_k(const float* __restrict__ x, float* __restrict__ y,
                           int HW) {
  const int bc = blockIdx.x;
  const float* p = x + (size_t)bc * HW;
  float* q = y + (size_t)bc * HW;
  __shared__ float s1[64], s2[64];
  float a = 0.0f, b = 0.0f;
  for (int i = threadIdx.x; i < HW; i += 64) {
    const float v = p[i];
    a += v;
    b += v * v;
  }
  s1[threadIdx.x] = a;
  s2[threadIdx.x] = b;
  __syncthreads();
  for (int o = 32; o > 0; o >>= 1) {
    if ((int)threadIdx.x < o) {
      s1[threadIdx.x] += s1[threadIdx.x + o];
      s2[threadIdx.x] += s2[threadIdx.x + o];
    }
    __syncthreads();
  }
  const float m   = s1[0] / (float)HW;
  const float var = s2[0] / (float)HW - m * m;
  const float r   = rsqrtf(var + 1e-5f);
  for (int i = threadIdx.x; i < HW; i += 64) q[i] = (p[i] - m) * r;
}
__global__ void binsel_k(const float* __restrict__ mem, const float* __restrict__ cur,
                         const int* __restrict__ hm, float* __restrict__ sel,
                         int total, int per_b) {
  int i = blockIdx.x * blockDim.x + threadIdx.x;
  if (i >= total) return;
  const int b = i / per_b;
  sel[i] = hm[b] ? mem[i] : cur[i];
}
__global__ void stepsel_k(const int* __restrict__ prog, int step,
                          const int* __restrict__ hm, const int* __restrict__ oset,
                          const float* __restrict__ old, const float* __restrict__ un,
                          const float* __restrict__ bn, const float* __restrict__ fk,
                          float* __restrict__ nxt, float* __restrict__ mem,
                          float* __restrict__ out_img, int total, int per_b) {
  int i = blockIdx.x * blockDim.x + threadIdx.x;
  if (i >= total) return;
  const int b = i / per_b;
  const int p = prog[b * PLENC + step];
  const float ov = old[i];
  float nv = ov;
  if (p >= PADC && p < PADC + NUC) nv = un[i];
  if (p >= PADC + NUC && hm[b])    nv = bn[i];
  if (p == 1) { nv = fk[i]; mem[i] = ov; }
  nxt[i] = nv;
  const int take = (!oset[b]) && (p == 0 || step == PLENC - 1);
  if (take) out_img[i] = nv;
}
__global__ void flags_k(const int* __restrict__ prog, int step, int* hm, int* oset) {
  const int b = threadIdx.x;
  if (b >= BATCH) return;
  const int p = prog[b * PLENC + step];
  const int take = (!oset[b]) && (p == 0 || step == PLENC - 1);
  oset[b] = oset[b] | take;
  hm[b]   = hm[b] | (p == 1);
}

// ---------------------------------------------------------------------------
extern "C" void kernel_launch(void* const* d_in, const int* in_sizes, int n_in,
                              void* d_out, int out_size, void* d_ws, size_t ws_size,
                              hipStream_t stream) {
  (void)in_sizes; (void)n_in; (void)out_size; (void)ws_size;
  const int* q    = (const int*)d_in[0];
  const int* prog = (const int*)d_in[1];
  const float* imgs = (const float*)d_in[2];
  auto in = [&](int i) { return (const float*)d_in[i]; };
  // params pytree flattened with sorted dict keys at every level:
  // 3..8  binary: W1,W2,W3,b1,b2,b3
  // 9..18 cls:    cW1,cW2,cW3,cb1,cb2,cb3,fW1,fW2,fb1,fb2
  // 19..28 ctrl_cnn (same order)   29 embed
  // 30..37 fork: W1..W4,b1..b4
  // 38..43 lstm: Whh0,Whh1,Wih0,Wih1,b0,b1
  // 44..45 proj: W,b
  // 46..51 rhn: Rh,Rt,Wh,Wt,bh,bt
  // 52..57 stem: W1,W2,W3,b1,b2,b3
  // 58..61 unary: W1,W2,b1,b2
  const float *binW1 = in(3), *binW2 = in(4), *binW3 = in(5);
  const float *binb1 = in(6), *binb2 = in(7), *binb3 = in(8);
  const float* clsP[10]  = {in(9),  in(10), in(11), in(12), in(13),
                            in(14), in(15), in(16), in(17), in(18)};
  const float* ctrlP[10] = {in(19), in(20), in(21), in(22), in(23),
                            in(24), in(25), in(26), in(27), in(28)};
  const float* embtab = in(29);
  const float *fkW1 = in(30), *fkW2 = in(31), *fkW3 = in(32), *fkW4 = in(33);
  const float *fkb1 = in(34), *fkb2 = in(35), *fkb3 = in(36), *fkb4 = in(37);
  const float *Whh0 = in(38), *Whh1 = in(39), *Wih0 = in(40), *Wih1 = in(41);
  const float *lb0 = in(42), *lb1 = in(43);
  const float *pW = in(44), *pb = in(45);
  const float *Rh = in(46), *Rt = in(47), *Wh = in(48), *Wtp = in(49);
  const float *bh = in(50), *bt = in(51);
  const float *sW1 = in(52), *sW2 = in(53), *sW3 = in(54);
  const float *sb1 = in(55), *sb2 = in(56), *sb3 = in(57);
  const float *uW1 = in(58), *uW2 = in(59), *ub1 = in(60), *ub2 = in(61);

  // ---------------- workspace carve-out (floats) ----------------
  float* wsf = (float*)d_ws;
  size_t off = 0;
  auto alloc = [&](size_t n) { float* p = wsf + off; off += n; return p; };
  const size_t IM = (size_t)BATCH * 32 * 196;   // 14x14x32 feature maps
  const size_t F6 = (size_t)BATCH * 192 * 196;  // fork 192-ch maps
  float* stem56 = alloc((size_t)BATCH * 32 * 56 * 56);
  float* stem28 = alloc((size_t)BATCH * 32 * 28 * 28);
  float* curA = alloc(IM);  float* curB = alloc(IM);
  float* orig = alloc(IM);  float* mem  = alloc(IM);
  float* oimg = alloc(IM);  float* fk   = alloc(IM);
  float* un   = alloc(IM);  float* bn   = alloc(IM);
  float* selb = alloc(IM);  float* u1   = alloc(IM);
  float* u2   = alloc(IM);  float* t1   = alloc(IM);
  float* t2   = alloc(IM);  float* b1b  = alloc(IM);
  float* b2b  = alloc(IM);
  float* f6a = alloc(F6); float* f6b = alloc(F6); float* f6c = alloc(F6);
  float* t3 = alloc((size_t)BATCH * 32 * 49);
  float* t4 = alloc((size_t)BATCH * 16 * 25);
  float* f1 = alloc((size_t)BATCH * 1024);
  float* feat = alloc((size_t)BATCH * 32);
  float* preh = alloc(4096); float* pret = alloc(4096); float* s = alloc(4096);
  float* hc = alloc(4 * 4096);  // h0,c0,h1,c1
  float *h0 = hc, *c0 = hc + 4096, *h1 = hc + 8192, *c1 = hc + 12288;
  float* z   = alloc((size_t)BATCH * 1024);
  float* ys  = alloc((size_t)20 * BATCH * 256);
  float* emb = alloc((size_t)BATCH * 20 * 128);
  float* flagsf = alloc(32);
  int* hm   = (int*)flagsf;
  int* oset = hm + 16;

  auto gemm = [&](const float* A, int lda, const float* W, const float* bias,
                  float* C, int ldc, int N, int K, int accum, int relu) {
    dim3 g((N + 63) / 64, 1, 1);
    gemm16_wmma<<<g, 128, 0, stream>>>(A, lda, W, bias, C, ldc, N, K, accum, relu);
  };
  auto conv = [&](const float* in0, const float* in1, int Cin0, int Cin,
                  const float* Wt, const float* Bs, const int* pr, int step,
                  int ioff, int imax, const float* res, float* out, int H, int Wd,
                  int OH, int OW, int Cout, int stride, int pad, int ksz, int relu) {
    dim3 g((OH * OW + 15) / 16, (Cout + 63) / 64, BATCH);
    conv_wmma<<<g, 128, 0, stream>>>(in0, in1, Cin0, Cin, Wt, Bs, pr, step, ioff,
                                     imax, res, out, H, Wd, OH, OW, Cout, stride,
                                     pad, ksz, relu);
  };
  auto zero = [&](float* p, int n) {
    zero_k<<<(n + 255) / 256, 256, 0, stream>>>(p, n);
  };
  auto copy = [&](const float* a, float* b, int n) {
    copy_k<<<(n + 255) / 256, 256, 0, stream>>>(a, b, n);
  };
  auto cnnfeats = [&](const float* x, const float* const* P, float* fcout, int ldc) {
    conv(x,  nullptr, 32, 32, P[0], P[3], nullptr, 0, 0, 0, nullptr, t1,
         14, 14, 14, 14, 32, 1, 1, 3, 1);                               // relu(conv1)
    conv(t1, nullptr, 32, 32, P[1], P[4], nullptr, 0, 0, 0, x, t2,
         14, 14, 14, 14, 32, 1, 1, 3, 1);                               // relu(conv2+x)
    maxpool2_k<<<(25088 + 255) / 256, 256, 0, stream>>>(t2, t3, 14, 14, 25088);
    conv(t3, nullptr, 32, 32, P[2], P[5], nullptr, 0, 0, 0, nullptr, t4,
         7, 7, 5, 5, 16, 1, 0, 3, 0);                                   // VALID conv3
    gemm(t4, 400, P[6], P[8], f1, 1024, 1024, 400, 0, 1);               // fc1 relu
    gemm(f1, 1024, P[7], P[9], fcout, ldc, 32, 1024, 0, 0);             // fc2
  };

  // -------- init (ws/out are poisoned by the harness) --------
  zero(hc, 4 * 4096);
  zero(mem, (int)IM);
  zero(oimg, (int)IM);
  zero(flagsf, 32);

  // -------- LSTM encoder --------
  embed_k<<<(BATCH * 20 * 128 + 255) / 256, 256, 0, stream>>>(q, embtab, emb);
  for (int t = 0; t < 20; ++t) {
    gemm(emb + (size_t)t * 128, 20 * 128, Wih0, lb0, z, 1024, 1024, 128, 0, 0);
    gemm(h0, 256, Whh0, nullptr, z, 1024, 1024, 256, 1, 0);
    lstm_gates_k<<<16, 256, 0, stream>>>(z, h0, c0, ys + (size_t)t * 4096);
  }
  for (int t = 0; t < 20; ++t) {
    gemm(ys + (size_t)t * 4096, 256, Wih1, lb1, z, 1024, 1024, 256, 0, 0);
    gemm(h1, 256, Whh1, nullptr, z, 1024, 1024, 256, 1, 0);
    lstm_gates_k<<<16, 256, 0, stream>>>(z, h1, c1, nullptr);
  }
  // cEnc == c1

  // -------- CNN stem (SAME, stride 2: pad_lo = 0, hi handled by bounds) ----
  conv(imgs,   nullptr, 3, 3,  sW1, sb1, nullptr, 0, 0, 0, nullptr, stem56,
       112, 112, 56, 56, 32, 2, 0, 3, 1);
  conv(stem56, nullptr, 32, 32, sW2, sb2, nullptr, 0, 0, 0, nullptr, stem28,
       56, 56, 28, 28, 32, 2, 0, 3, 1);
  conv(stem28, nullptr, 32, 32, sW3, sb3, nullptr, 0, 0, 0, nullptr, curA,
       28, 28, 14, 14, 32, 2, 0, 3, 1);
  copy(curA, orig, (int)IM);

  float* cur = curA;
  float* nxt = curB;
  float* dout = (float*)d_out;

  // -------- program loop --------
  for (int i = 0; i < PLENC; ++i) {
    // control features + RHN + projection
    cnnfeats(cur, ctrlP, feat, 32);
    copy(c1, s, 4096);
    // layer 0: feat @ Wh/Wt + s @ Rh0/Rt0 + b
    gemm(feat, 32, Wh,  bh, preh, 256, 256, 32, 0, 0);
    gemm(s, 256, Rh, nullptr, preh, 256, 256, 256, 1, 0);
    gemm(feat, 32, Wtp, bt, pret, 256, 256, 32, 0, 0);
    gemm(s, 256, Rt, nullptr, pret, 256, 256, 256, 1, 0);
    rhn_pw_k<<<16, 256, 0, stream>>>(preh, pret, s);
    for (int l = 1; l < 3; ++l) {
      gemm(s, 256, Rh + (size_t)l * 65536, bh + (size_t)l * 256, preh, 256, 256, 256, 0, 0);
      gemm(s, 256, Rt + (size_t)l * 65536, bt + (size_t)l * 256, pret, 256, 256, 256, 0, 0);
      rhn_pw_k<<<16, 256, 0, stream>>>(preh, pret, s);
    }
    gemm(s, 256, pW, pb, dout + (size_t)i * 43, PLENC * 43, 43, 256, 0, 0);

    // fork path: concat(cur,orig) -> 1x1 -> 3x3 -> 3x3(+res) -> 1x1
    conv(cur, orig, 32, 64, fkW1, fkb1, nullptr, 0, 0, 0, nullptr, f6a,
         14, 14, 14, 14, 192, 1, 0, 1, 1);
    conv(f6a, nullptr, 192, 192, fkW2, fkb2, nullptr, 0, 0, 0, nullptr, f6b,
         14, 14, 14, 14, 192, 1, 1, 3, 1);
    conv(f6b, nullptr, 192, 192, fkW3, fkb3, nullptr, 0, 0, 0, f6a, f6c,
         14, 14, 14, 14, 192, 1, 1, 3, 1);
    conv(f6c, nullptr, 192, 192, fkW4, fkb4, nullptr, 0, 0, 0, nullptr, fk,
         14, 14, 14, 14, 32, 1, 0, 1, 0);

    // unary path (per-sample weight bank clip(p-PAD, 0, NU-1))
    conv(cur, nullptr, 32, 32, uW1, ub1, prog, i, PADC, NUC - 1, nullptr, u1,
         14, 14, 14, 14, 32, 1, 1, 3, 1);
    conv(u1, nullptr, 32, 32, uW2, ub2, prog, i, PADC, NUC - 1, cur, u2,
         14, 14, 14, 14, 32, 1, 1, 3, 1);
    instnorm_k<<<BATCH * 32, 64, 0, stream>>>(u2, un, 196);

    // binary path (first operand: has_mem ? mem : cur)
    binsel_k<<<((int)IM + 255) / 256, 256, 0, stream>>>(mem, cur, hm, selb,
                                                        (int)IM, 32 * 196);
    conv(selb, cur, 32, 64, binW1, binb1, prog, i, PADC + NUC, NBC - 1, nullptr,
         b1b, 14, 14, 14, 14, 32, 1, 0, 1, 1);
    conv(b1b, nullptr, 32, 32, binW2, binb2, prog, i, PADC + NUC, NBC - 1, nullptr,
         b2b, 14, 14, 14, 14, 32, 1, 1, 3, 1);
    conv(b2b, nullptr, 32, 32, binW3, binb3, prog, i, PADC + NUC, NBC - 1, b1b,
         bn, 14, 14, 14, 14, 32, 1, 1, 3, 1);

    // per-sample op selection + memory/out_img update, then flag update
    stepsel_k<<<((int)IM + 255) / 256, 256, 0, stream>>>(
        prog, i, hm, oset, cur, un, bn, fk, nxt, mem, oimg, (int)IM, 32 * 196);
    flags_k<<<1, 16, 0, stream>>>(prog, i, hm, oset);
    float* tmp = cur; cur = nxt; nxt = tmp;
  }

  // -------- classifier head --------
  cnnfeats(oimg, clsP, dout + 4128, 32);
}
